// PNNStack_9148280341024
// MI455X (gfx1250) — compile-verified, matrix operand order checked by reading, not run on hardware
//
#include <hip/hip_runtime.h>
#include <hip/hip_bf16.h>

// ---------------------------------------------------------------------------
// PNA GNN stack for MI455X (gfx1250, wave32, WMMA).
// Dense transforms -> v_wmma_f32_16x16x32_bf16 with bf16-padded weight tiles;
// edge scatter -> fp32 atomics (+ order-preserving u32 min/max atomics);
// async-to-LDS staging where the toolchain exposes it.
// ---------------------------------------------------------------------------

#define Nn 20000
#define Ee 320000
#define FIc 16
#define Hh 100
#define Tt 5
#define FOo 20
#define Bb 50
#define EPB 8   // edges per block in edge_reduce

typedef __bf16 bf16;
typedef __attribute__((ext_vector_type(16))) __bf16 v16bf;
typedef __attribute__((ext_vector_type(8)))  float  v8f;
typedef __attribute__((ext_vector_type(8)))  unsigned short v8s;
typedef __attribute__((ext_vector_type(16))) unsigned short v16s;
typedef __attribute__((ext_vector_type(4)))  int v4i;

#if defined(__has_builtin)
# if __has_builtin(__builtin_amdgcn_global_load_async_to_lds_b128)
#   define HAVE_ASYNC_B128 1
# endif
# if __has_builtin(__builtin_amdgcn_s_wait_asynccnt)
#   define HAVE_WAIT_ASYNC 1
# endif
#endif

// ---------------- small helpers ----------------

__device__ __forceinline__ unsigned short f2bf(float v) {
    // truncating f32 -> bf16 (fast; fine for GEMM inputs)
    return (unsigned short)(__float_as_uint(v) >> 16);
}

// order-preserving float<->uint encoding for single-instruction atomic min/max
__device__ __forceinline__ unsigned encF(float f) {
    unsigned u = __float_as_uint(f);
    return (u & 0x80000000u) ? ~u : (u | 0x80000000u);
}
__device__ __forceinline__ float decF(unsigned e) {
    unsigned u = (e & 0x80000000u) ? (e & 0x7fffffffu) : ~e;
    return __uint_as_float(u);
}

__global__ void fill_u32(unsigned* __restrict__ p, unsigned v, size_t n) {
    size_t i = (size_t)blockIdx.x * blockDim.x + threadIdx.x;
    if (i < n) p[i] = v;
}

// B (KxNc, f32) -> Bt (Ncpad x Kpad, bf16, transposed, zero padded)
__global__ void convert_bt(const float* __restrict__ B, int ldb, int K, int Nc,
                           unsigned short* __restrict__ Bt, int Kpad, int Ncpad) {
    int idx = blockIdx.x * blockDim.x + threadIdx.x;
    if (idx >= Ncpad * Kpad) return;
    int n = idx / Kpad, k = idx % Kpad;
    float v = (n < Nc && k < K) ? B[(size_t)k * ldb + n] : 0.0f;
    Bt[idx] = f2bf(v);
}

// ---------------- generic bf16 WMMA GEMM: C = A(MxK) * Bt (+bias)(+C)(relu) -----

__global__ __launch_bounds__(128)
void wmma_gemm_kernel(const float* __restrict__ A, int lda,
                      const unsigned short* __restrict__ Bt, int ldbt, // bf16 [Ncpad][Kpad]
                      const float* __restrict__ bias,
                      float* __restrict__ C, int ldc,
                      int M, int Nc, int K, int accum, int relu)
{
    // 64x64 output tile per block, 4 waves, each wave: 16 rows x 64 cols (4 WMMA tiles)
    __shared__ alignas(16) unsigned short As[64][40];  // [row][k] bf16, row stride 80B
    __shared__ alignas(16) unsigned short Bs[64][40];  // [n][k]   bf16 (pre-transposed)

    const int tid  = threadIdx.x;
    const int lane = tid & 31;
    const int wave = tid >> 5;
    const int rowBase = blockIdx.y * 64;
    const int colBase = blockIdx.x * 64;

    const int half = lane >> 4;       // 0: lanes 0-15, 1: lanes 16-31
    const int l16  = lane & 15;
    const int kb   = half ? 8 : 0;    // ISA 16-bit A/B fragment K base per half-wave
    const bool aVec = ((lda & 3) == 0);

    v8f acc[4];
    #pragma unroll
    for (int nt = 0; nt < 4; ++nt) { v8f z = {}; acc[nt] = z; }

    const int kTiles = (K + 31) >> 5;
    for (int kt = 0; kt < kTiles; ++kt) {
        const int k0 = kt << 5;
        // ---- stage A tile (64x32) as bf16: 512 chunks of 4 elements ----
        #pragma unroll
        for (int i = 0; i < 4; ++i) {
            int chunk = i * 128 + tid;
            int r = chunk >> 3;
            int c = (chunk & 7) << 2;
            int gr = rowBase + r;
            int gc = k0 + c;
            float x0 = 0.f, x1 = 0.f, x2 = 0.f, x3 = 0.f;
            if (aVec && gr < M && gc + 3 < K) {
                float4 t = *(const float4*)(A + (size_t)gr * lda + gc);
                x0 = t.x; x1 = t.y; x2 = t.z; x3 = t.w;
            } else if (gr < M) {
                const float* ap = A + (size_t)gr * lda;
                if (gc     < K) x0 = ap[gc];
                if (gc + 1 < K) x1 = ap[gc + 1];
                if (gc + 2 < K) x2 = ap[gc + 2];
                if (gc + 3 < K) x3 = ap[gc + 3];
            }
            unsigned lo = ((unsigned)f2bf(x1) << 16) | f2bf(x0);
            unsigned hi = ((unsigned)f2bf(x3) << 16) | f2bf(x2);
            *(uint2*)&As[r][c] = make_uint2(lo, hi);
        }
        // ---- stage B tile (64 n x 32 k) bf16: unguarded (Bt is padded) ----
        #pragma unroll
        for (int i = 0; i < 2; ++i) {
            int chunk = i * 128 + tid;        // 0..255 chunks of 8 bf16
            int nn = chunk >> 2;
            int c = (chunk & 3) << 3;
            v8s t = *(const v8s*)(Bt + (size_t)(colBase + nn) * ldbt + (k0 + c));
            *(v8s*)&Bs[nn][c] = t;
        }
        __syncthreads();

        // A fragment: lanes<16: K 0-7 & 16-23; lanes>=16: K 8-15 & 24-31
        v16bf afrag;
        {
            v8s lo = *(const v8s*)&As[wave * 16 + l16][kb];
            v8s hi = *(const v8s*)&As[wave * 16 + l16][kb + 16];
            v16s av;
            #pragma unroll
            for (int i = 0; i < 8; ++i) { av[i] = lo[i]; av[8 + i] = hi[i]; }
            afrag = __builtin_bit_cast(v16bf, av);
        }
        #pragma unroll
        for (int nt = 0; nt < 4; ++nt) {
            v16bf bfrag;
            v8s lo = *(const v8s*)&Bs[nt * 16 + l16][kb];
            v8s hi = *(const v8s*)&Bs[nt * 16 + l16][kb + 16];
            v16s bv;
            #pragma unroll
            for (int i = 0; i < 8; ++i) { bv[i] = lo[i]; bv[8 + i] = hi[i]; }
            bfrag = __builtin_bit_cast(v16bf, bv);
            acc[nt] = __builtin_amdgcn_wmma_f32_16x16x32_bf16(
                false, afrag, false, bfrag, (short)0, acc[nt], false, false);
        }
        __syncthreads();
    }

    // store: C/D layout -> VGPR r: row = r + 8*half; col = lane&15
    #pragma unroll
    for (int nt = 0; nt < 4; ++nt) {
        int n = colBase + nt * 16 + l16;
        if (n >= Nc) continue;
        #pragma unroll
        for (int r = 0; r < 8; ++r) {
            int m = rowBase + wave * 16 + r + half * 8;
            if (m >= M) continue;
            float v = acc[nt][r];
            if (bias) v += bias[n];
            float* cp = &C[(size_t)m * ldc + n];
            if (accum) v += *cp;
            if (relu) v = fmaxf(v, 0.0f);
            *cp = v;
        }
    }
}

// ---------------- graph statistics ----------------

__global__ void degree_kernel(const int* __restrict__ dst, float* __restrict__ cnt, int e_) {
    int e = blockIdx.x * blockDim.x + threadIdx.x;
    if (e < e_) atomicAdd(&cnt[dst[e]], 1.0f);
}

__global__ void stats_kernel(const float* __restrict__ cnt, float* __restrict__ stats, int n_) {
    __shared__ float s1[256], s2[256];
    int tid = threadIdx.x;
    float a = 0.f, b = 0.f;
    for (int i = blockIdx.x * 256 + tid; i < n_; i += gridDim.x * 256) {
        float c = cnt[i]; a += c; b += logf(c + 1.0f);
    }
    s1[tid] = a; s2[tid] = b; __syncthreads();
    for (int st = 128; st > 0; st >>= 1) {
        if (tid < st) { s1[tid] += s1[tid + st]; s2[tid] += s2[tid + st]; }
        __syncthreads();
    }
    if (tid == 0) { atomicAdd(&stats[0], s1[0]); atomicAdd(&stats[1], s2[0]); }
}

__global__ void compute_scal(const float* __restrict__ cnt, const float* __restrict__ stats,
                             float* __restrict__ scal, int n_) {
    int n = blockIdx.x * blockDim.x + threadIdx.x;
    if (n >= n_) return;
    float c = cnt[n];
    float d = fmaxf(c, 1.0f);
    float ld = logf(d + 1.0f);
    float avg_lin = stats[0] / (float)n_;
    float avg_log = stats[1] / (float)n_;
    scal[n]            = 1.0f;
    scal[n_ + n]       = ld / avg_log;
    scal[2 * n_ + n]   = avg_log / ld;
    scal[3 * n_ + n]   = d / avg_lin;
}

// ---------------- weight repacking ----------------

__global__ void repack_pre_w(const float* __restrict__ pw, float* __restrict__ Wi,
                             float* __restrict__ Wj, int f, int TF) {
    int idx = blockIdx.x * blockDim.x + threadIdx.x;
    if (idx >= f * TF) return;
    int c = idx / TF, tf = idx % TF, t = tf / f, g = tf % f;
    const float* base = pw + (size_t)t * 3 * f * f;
    Wi[idx] = base[(size_t)c * f + g];
    Wj[idx] = base[((size_t)f + c) * f + g];
}

__global__ void compute_uc(const float* __restrict__ pw, const float* __restrict__ ew,
                           const float* __restrict__ eb, const float* __restrict__ pb,
                           float* __restrict__ u, float* __restrict__ cvec, int f, int TF) {
    int tf = blockIdx.x * blockDim.x + threadIdx.x;
    if (tf >= TF) return;
    int t = tf / f, g = tf % f;
    const float* we = pw + ((size_t)t * 3 * f + 2 * f) * f;
    float su = 0.f, sv = 0.f;
    for (int c = 0; c < f; ++c) {
        float w = we[(size_t)c * f + g];
        su += ew[c] * w; sv += eb[c] * w;
    }
    u[tf] = su;
    cvec[tf] = sv + pb[tf];
}

__global__ void repack_wx(const float* __restrict__ qw, float* __restrict__ Wx,
                          int f, int FO, int T) {
    int idx = blockIdx.x * blockDim.x + threadIdx.x;
    if (idx >= f * T * FO) return;
    int c = idx / (T * FO), r = idx % (T * FO), t = r / FO, g = r % FO;
    Wx[idx] = qw[((size_t)t * 17 * f + c) * FO + g];
}

__global__ void repack_wcat(const float* __restrict__ qw, float* __restrict__ Wcat,
                            int f, int FO, int T) {
    int idx = blockIdx.x * blockDim.x + threadIdx.x;
    int tot = T * 4 * f * 4 * FO;
    if (idx >= tot) return;
    int t = idx / (4 * f * 4 * FO);
    int rem = idx % (4 * f * 4 * FO);
    int c = rem / (4 * FO);
    int sg = rem % (4 * FO);
    int s = sg / FO, g = sg % FO;
    Wcat[idx] = qw[((size_t)t * 17 * f + f + (size_t)s * 4 * f + c) * FO + g];
}

// ---------------- edge scatter: y = B[src] + a*u + c ; segment sum/sq/min/max ----

__global__ __launch_bounds__(128)
void edge_reduce_kernel(const int* __restrict__ src, const int* __restrict__ dst,
                        const float* __restrict__ ea, const float* __restrict__ ysrc,
                        const float* __restrict__ u, const float* __restrict__ cvec,
                        float* __restrict__ rsum, float* __restrict__ rsq,
                        unsigned* __restrict__ rmin, unsigned* __restrict__ rmax,
                        int n_, int f, int TF)
{
    __shared__ alignas(16) float sU[512];
    __shared__ alignas(16) float sC[512];
    // stage u, cvec (TF is a multiple of 4: 80 or 500)
#ifdef HAVE_ASYNC_B128
    {
        int tf4 = TF >> 2;
        for (int c = threadIdx.x; c < tf4; c += 128) {
            __builtin_amdgcn_global_load_async_to_lds_b128(
                (__attribute__((address_space(1))) v4i*)(void*)(u + c * 4),
                (__attribute__((address_space(3))) v4i*)&sU[c * 4], 0, 0);
            __builtin_amdgcn_global_load_async_to_lds_b128(
                (__attribute__((address_space(1))) v4i*)(void*)(cvec + c * 4),
                (__attribute__((address_space(3))) v4i*)&sC[c * 4], 0, 0);
        }
#ifdef HAVE_WAIT_ASYNC
        __builtin_amdgcn_s_wait_asynccnt(0);
#else
        asm volatile("s_wait_asynccnt 0x0" ::: "memory");
#endif
    }
#else
    for (int c = threadIdx.x; c < TF; c += 128) { sU[c] = u[c]; sC[c] = cvec[c]; }
#endif
    __syncthreads();

    int e0 = blockIdx.x * EPB;
    int e1 = e0 + EPB; if (e1 > Ee) e1 = Ee;
    for (int e = e0; e < e1; ++e) {
        if (e + 1 < e1)
            __builtin_prefetch(ysrc + (size_t)src[e + 1] * TF, 0, 1);
        int s = src[e], d = dst[e];
        float a = ea[e];
        const float* yrow = ysrc + (size_t)s * TF;
        for (int tf = threadIdx.x; tf < TF; tf += 128) {
            int t = tf / f, j = tf - t * f;
            float y = yrow[tf] + a * sU[tf] + sC[tf];
            size_t ridx = ((size_t)t * n_ + d) * f + j;   // component layout [t][n][f]
            atomicAdd(&rsum[ridx], y);
            atomicAdd(&rsq[ridx], y * y);
            unsigned enc = encF(y);
            atomicMax(&rmax[ridx], enc);
            atomicMin(&rmin[ridx], enc);
        }
    }
}

// finalize: mean/min/max = A + agg(y) (or 0 if no edges); std from y stats; in-place
__global__ void finalize_agg(const float* __restrict__ cnt, const float* __restrict__ xw_dst,
                             float* __restrict__ rsum, float* __restrict__ rsq,
                             float* __restrict__ rminb, float* __restrict__ rmaxb,
                             int n_, int f, int TF)
{
    int idx = blockIdx.x * blockDim.x + threadIdx.x;
    if (idx >= n_ * TF) return;
    int n = idx / TF, tf = idx % TF, t = tf / f, j = tf % f;
    size_t ridx = ((size_t)t * n_ + n) * f + j;
    float c = cnt[n];
    float d = fmaxf(c, 1.0f);
    float s1 = rsum[ridx], s2 = rsq[ridx];
    float mean_y = s1 / d;
    float var = s2 / d - mean_y * mean_y;
    float stdv = sqrtf(fmaxf(var, 0.0f) + 1e-5f);
    float mean, mn, mx;
    if (c > 0.0f) {
        float A = xw_dst[(size_t)n * TF + tf];
        mean = A + mean_y;
        mn = A + decF(((const unsigned*)rminb)[ridx]);
        mx = A + decF(((const unsigned*)rmaxb)[ridx]);
    } else {
        mean = 0.0f; mn = 0.0f; mx = 0.0f;
    }
    rsum[ridx] = mean; rminb[ridx] = mn; rmaxb[ridx] = mx; rsq[ridx] = stdv;
}

// h[n][t*FO+g] = xwx + sum_s scal[s][n] * Z[n][t*4FO + s*FO + g] + qb[t*FO+g]
__global__ void combine_post(const float* __restrict__ xwx, const float* __restrict__ Z,
                             const float* __restrict__ scal, const float* __restrict__ qb,
                             float* __restrict__ hpost, int n_, int T_, int FO_)
{
    int idx = blockIdx.x * blockDim.x + threadIdx.x;
    int TFO = T_ * FO_;
    if (idx >= n_ * TFO) return;
    int n = idx / TFO, r = idx % TFO, t = r / FO_, g = r % FO_;
    const float* zrow = Z + (size_t)n * (T_ * 4 * FO_) + t * 4 * FO_;
    float v = xwx[idx] + qb[r]
            + scal[n]            * zrow[g]
            + scal[n_ + n]       * zrow[FO_ + g]
            + scal[2 * n_ + n]   * zrow[2 * FO_ + g]
            + scal[3 * n_ + n]   * zrow[3 * FO_ + g];
    hpost[idx] = v;
}

// ---------------- batch norm ----------------

__global__ void bn_stats(const float* __restrict__ h, float* __restrict__ stat, int n_, int H_) {
    __shared__ float s1[256], s2[256];
    int col = blockIdx.x, tid = threadIdx.x;
    float a = 0.f, b = 0.f;
    for (int i = tid; i < n_; i += 256) {
        float v = h[(size_t)i * H_ + col]; a += v; b += v * v;
    }
    s1[tid] = a; s2[tid] = b; __syncthreads();
    for (int st = 128; st > 0; st >>= 1) {
        if (tid < st) { s1[tid] += s1[tid + st]; s2[tid] += s2[tid + st]; }
        __syncthreads();
    }
    if (tid == 0) {
        float mu = s1[0] / (float)n_;
        float var = s2[0] / (float)n_ - mu * mu;
        stat[col] = mu;
        stat[H_ + col] = rsqrtf(fmaxf(var, 0.0f) + 1e-5f);
    }
}

__global__ void bn_apply(float* __restrict__ h, const float* __restrict__ stat,
                         const float* __restrict__ g, const float* __restrict__ b,
                         int n_, int H_) {
    int idx = blockIdx.x * blockDim.x + threadIdx.x;
    if (idx >= n_ * H_) return;
    int col = idx % H_;
    float v = (h[idx] - stat[col]) * stat[H_ + col] * g[col] + b[col];
    h[idx] = fmaxf(v, 0.0f);
}

// ---------------- pooling ----------------

__global__ void pool_kernel(const float* __restrict__ h, const int* __restrict__ batch,
                            float* __restrict__ gsum, float* __restrict__ gcnt, int n_, int H_) {
    int idx = blockIdx.x * blockDim.x + threadIdx.x;
    if (idx >= n_ * H_) return;
    int n = idx / H_, c = idx % H_;
    int bb = batch[n];
    atomicAdd(&gsum[(size_t)bb * H_ + c], h[idx]);
    if (c == 0) atomicAdd(&gcnt[bb], 1.0f);
}

__global__ void pool_avg(float* __restrict__ gsum, const float* __restrict__ gcnt, int B_, int H_) {
    int idx = blockIdx.x * blockDim.x + threadIdx.x;
    if (idx >= B_ * H_) return;
    gsum[idx] /= fmaxf(gcnt[idx / H_], 1.0f);
}

// ---------------- host-side helpers ----------------

static void gemm(hipStream_t st, const float* A, int lda, const float* B, int ldb,
                 const float* bias, float* C, int ldc, int M, int Nc, int K,
                 int accum, int relu, unsigned short* Bt) {
    int Kpad  = ((K + 31) / 32) * 32;
    int Ncpad = ((Nc + 63) / 64) * 64;
    int tot = Ncpad * Kpad;
    convert_bt<<<(tot + 255) / 256, 256, 0, st>>>(B, ldb, K, Nc, Bt, Kpad, Ncpad);
    dim3 grid((unsigned)(Ncpad / 64), (unsigned)((M + 63) / 64));
    wmma_gemm_kernel<<<grid, 128, 0, st>>>(A, lda, Bt, Kpad, bias, C, ldc, M, Nc, K, accum, relu);
}

static void fillu(hipStream_t st, float* p, unsigned v, size_t n) {
    fill_u32<<<(unsigned)((n + 255) / 256), 256, 0, st>>>((unsigned*)p, v, n);
}

extern "C" void kernel_launch(void* const* d_in, const int* in_sizes, int n_in,
                              void* d_out, int out_size, void* d_ws, size_t ws_size,
                              hipStream_t stream) {
    (void)in_sizes; (void)n_in; (void)out_size; (void)ws_size;

    const float* x        = (const float*)d_in[0];
    const int*   eidx     = (const int*)d_in[1];   // (2,E)
    const float* eattr    = (const float*)d_in[2]; // (E,1)
    const int*   batch    = (const int*)d_in[3];
    const float* e0_w     = (const float*)d_in[4];
    const float* e0_b     = (const float*)d_in[5];
    const float* pre0_w   = (const float*)d_in[6];
    const float* pre0_b   = (const float*)d_in[7];
    const float* post0_w  = (const float*)d_in[8];
    const float* post0_b  = (const float*)d_in[9];
    const float* lin0_w   = (const float*)d_in[10];
    const float* lin0_b   = (const float*)d_in[11];
    const float* bn0_g    = (const float*)d_in[12];
    const float* bn0_b    = (const float*)d_in[13];
    const float* eL_w     = (const float*)d_in[14];
    const float* eL_b     = (const float*)d_in[15];
    const float* preL_w   = (const float*)d_in[16];
    const float* preL_b   = (const float*)d_in[17];
    const float* postL_w  = (const float*)d_in[18];
    const float* postL_b  = (const float*)d_in[19];
    const float* linL_w   = (const float*)d_in[20];
    const float* linL_b   = (const float*)d_in[21];
    const float* bnL_g    = (const float*)d_in[22];
    const float* bnL_b    = (const float*)d_in[23];
    const float* mlp_w1   = (const float*)d_in[24];
    const float* mlp_b1   = (const float*)d_in[25];
    const float* mlp_w2   = (const float*)d_in[26];
    const float* mlp_b2   = (const float*)d_in[27];
    const float* mlp_w3   = (const float*)d_in[28];
    const float* mlp_b3   = (const float*)d_in[29];

    const int* srcI = eidx;
    const int* dstI = eidx + Ee;

    // ---- workspace layout (floats, 16B-aligned allocations) ----
    float* W = (float*)d_ws;
    size_t o = 0;
    auto F = [&](size_t n) { n = (n + 3) & ~(size_t)3; float* p = W + o; o += n; return p; };
    float* cnt    = F(Nn);
    float* stats  = F(8);
    float* scal   = F(4 * (size_t)Nn);
    float* uvec   = F(Tt * Hh);
    float* cvec   = F(Tt * Hh);
    float* Wi     = F((size_t)Hh * Tt * Hh);
    float* Wj     = F((size_t)Hh * Tt * Hh);
    float* Wx     = F((size_t)Hh * Tt * FOo);
    float* Wcat   = F((size_t)Tt * 4 * Hh * 4 * FOo);
    float* Btbuf  = F(512 * 128 / 2);                  // bf16 scratch: 512x128 ushorts
    float* xw_dst = F((size_t)Nn * Tt * Hh);
    float* xw_src = F((size_t)Nn * Tt * Hh);
    float* xwx    = F((size_t)Nn * Tt * FOo);
    float* rsum   = F((size_t)Nn * Tt * Hh);
    float* rsq    = F((size_t)Nn * Tt * Hh);
    float* rminb  = F((size_t)Nn * Tt * Hh);
    float* rmaxb  = F((size_t)Nn * Tt * Hh);
    float* Z      = F((size_t)Nn * Tt * 4 * FOo);
    float* hpost  = F((size_t)Nn * Tt * FOo);
    float* hA     = F((size_t)Nn * Hh);
    float* bnstat = F(2 * Hh);
    float* gsum   = F((size_t)Bb * Hh);
    float* gcnt   = F(Bb);
    float* m1     = F((size_t)Bb * 50);
    float* m2     = F((size_t)Bb * 25);
    unsigned short* Bt = (unsigned short*)Btbuf;

    // ---- graph statistics (once) ----
    fillu(stream, cnt, 0u, Nn);
    fillu(stream, stats, 0u, 8);
    degree_kernel<<<(Ee + 255) / 256, 256, 0, stream>>>(dstI, cnt, Ee);
    stats_kernel<<<64, 256, 0, stream>>>(cnt, stats, Nn);
    compute_scal<<<(Nn + 255) / 256, 256, 0, stream>>>(cnt, stats, scal, Nn);

    // ---- one PNA conv layer ----
    auto run_conv = [&](int f, const float* xin,
                        const float* ew, const float* eb,
                        const float* pw, const float* pb,
                        const float* qw, const float* qb,
                        const float* lw, const float* lb) {
        const int TF  = Tt * f;
        const int TFO = Tt * FOo;
        // weight repacks
        repack_pre_w<<<(f * TF + 255) / 256, 256, 0, stream>>>(pw, Wi, Wj, f, TF);
        compute_uc<<<(TF + 255) / 256, 256, 0, stream>>>(pw, ew, eb, pb, uvec, cvec, f, TF);
        repack_wx<<<(f * TFO + 255) / 256, 256, 0, stream>>>(qw, Wx, f, FOo, Tt);
        repack_wcat<<<(Tt * 4 * f * 4 * FOo + 255) / 256, 256, 0, stream>>>(qw, Wcat, f, FOo, Tt);
        // node-level pre-transforms (WMMA)
        gemm(stream, xin, f, Wi, TF, nullptr, xw_dst, TF, Nn, TF, f, 0, 0, Bt);
        gemm(stream, xin, f, Wj, TF, nullptr, xw_src, TF, Nn, TF, f, 0, 0, Bt);
        gemm(stream, xin, f, Wx, TFO, nullptr, xwx, TFO, Nn, TFO, f, 0, 0, Bt);
        // reset segment reduction buffers
        size_t nr = (size_t)Nn * TF;
        fillu(stream, rsum, 0u, nr);
        fillu(stream, rsq, 0u, nr);
        fillu(stream, rminb, 0xFFFFFFFFu, nr);
        fillu(stream, rmaxb, 0u, nr);
        // per-edge scatter
        edge_reduce_kernel<<<(Ee + EPB - 1) / EPB, 128, 0, stream>>>(
            srcI, dstI, eattr, xw_src, uvec, cvec,
            rsum, rsq, (unsigned*)rminb, (unsigned*)rmaxb, Nn, f, TF);
        // finalize aggregations (in place: rsum=mean, rminb=min, rmaxb=max, rsq=std)
        finalize_agg<<<((size_t)Nn * TF + 255) / 256, 256, 0, stream>>>(
            cnt, xw_dst, rsum, rsq, rminb, rmaxb, Nn, f, TF);
        // aggregation GEMMs: Z[n][t*4FO + s*FO + g], accumulated over 4 components
        fillu(stream, Z, 0u, (size_t)Nn * Tt * 4 * FOo);
        const float* comp[4] = { rsum, rminb, rmaxb, rsq };
        for (int t = 0; t < Tt; ++t)
            for (int c4 = 0; c4 < 4; ++c4)
                gemm(stream,
                     comp[c4] + (size_t)t * Nn * f, f,
                     Wcat + ((size_t)t * 4 * f + (size_t)c4 * f) * (4 * FOo), 4 * FOo,
                     nullptr,
                     Z + t * 4 * FOo, Tt * 4 * FOo,
                     Nn, 4 * FOo, f, /*accum=*/1, 0, Bt);
        // scaler mixing + bias
        combine_post<<<((size_t)Nn * TFO + 255) / 256, 256, 0, stream>>>(
            xwx, Z, scal, qb, hpost, Nn, Tt, FOo);
        // final linear (WMMA)
        gemm(stream, hpost, TFO, lw, Hh, lb, hA, Hh, Nn, Hh, TFO, 0, 0, Bt);
    };

    auto run_bn = [&](const float* g, const float* b) {
        bn_stats<<<Hh, 256, 0, stream>>>(hA, bnstat, Nn, Hh);
        bn_apply<<<((size_t)Nn * Hh + 255) / 256, 256, 0, stream>>>(hA, bnstat, g, b, Nn, Hh);
    };

    // layer 0 (f = 16)
    run_conv(FIc, x, e0_w, e0_b, pre0_w, pre0_b, post0_w, post0_b, lin0_w, lin0_b);
    run_bn(bn0_g, bn0_b);

    // layers 1,2 (f = 100)
    for (int i = 0; i < 2; ++i) {
        run_conv(Hh, hA,
                 eL_w + (size_t)i * Hh, eL_b + (size_t)i * Hh,
                 preL_w + (size_t)i * Tt * 3 * Hh * Hh, preL_b + (size_t)i * Tt * Hh,
                 postL_w + (size_t)i * Tt * 17 * Hh * FOo, postL_b + (size_t)i * Tt * FOo,
                 linL_w + (size_t)i * Hh * Hh, linL_b + (size_t)i * Hh);
        run_bn(bnL_g + (size_t)i * Hh, bnL_b + (size_t)i * Hh);
    }

    // pooling + MLP head (WMMA)
    fillu(stream, gsum, 0u, (size_t)Bb * Hh);
    fillu(stream, gcnt, 0u, Bb);
    pool_kernel<<<((size_t)Nn * Hh + 255) / 256, 256, 0, stream>>>(hA, batch, gsum, gcnt, Nn, Hh);
    pool_avg<<<(Bb * Hh + 255) / 256, 256, 0, stream>>>(gsum, gcnt, Bb, Hh);
    gemm(stream, gsum, Hh, mlp_w1, 50, mlp_b1, m1, 50, Bb, 50, Hh, 0, 1, Bt);
    gemm(stream, m1, 50, mlp_w2, 25, mlp_b2, m2, 25, Bb, 25, 50, 0, 1, Bt);
    gemm(stream, m2, 25, mlp_w3, 1, mlp_b3, (float*)d_out, 1, Bb, 1, 25, 0, 0, Bt);
}
